// TFSwinTransformerBlock3D_46033459479034
// MI455X (gfx1250) — matrix operand order repaired and checked by reading, not compiled, not run on hardware
//
#include <hip/hip_runtime.h>
#include <hip/hip_bf16.h>

typedef unsigned int u32;
typedef unsigned short u16;
typedef __attribute__((ext_vector_type(16))) __bf16 v16bf;
typedef __attribute__((ext_vector_type(8)))  float  v8f;

union Frag { v16bf v; u32 u[8]; };

#define NWIN  128
#define NTOK  392
#define NPAD  416
#define CDIM  128
#define NHEAD 4
#define HDIM  32
#define HIDD  512
#define MTOT  (NWIN*NTOK)   // 50176

__device__ __forceinline__ u16 f2bf(float f){
  u32 u = __float_as_uint(f);
  return (u16)((u + 0x7fffu + ((u >> 16) & 1u)) >> 16);   // RNE
}

__device__ __forceinline__ v8f vzero(){
  v8f z;
  #pragma unroll
  for (int i = 0; i < 8; ++i) z[i] = 0.0f;
  return z;
}

__device__ __forceinline__ v8f wmma_bf16(const Frag& a, const Frag& b, v8f c){
  return __builtin_amdgcn_wmma_f32_16x16x32_bf16(false, a.v, false, b.v,
                                                 (short)0, c, false, false);
}

// ---------------------------------------------------------------- utilities
__global__ void swin_zero_u32(u32* p, long n){
  long i = (long)blockIdx.x * blockDim.x + threadIdx.x;
  long st = (long)gridDim.x * blockDim.x;
  for (; i < n; i += st) p[i] = 0u;
}

__global__ void swin_cvt_bf16(const float* __restrict__ src, u16* __restrict__ dst, long n){
  long i = (long)blockIdx.x * blockDim.x + threadIdx.x;
  long st = (long)gridDim.x * blockDim.x;
  for (; i < n; i += st) dst[i] = f2bf(src[i]);
}

// bias[h][i][j] = rel_table[rel_index[i][j]][h]
__global__ void swin_bias_kernel(const float* __restrict__ tbl, const int* __restrict__ ridx,
                                 float* __restrict__ bias){
  int i = blockIdx.x * blockDim.x + threadIdx.x;
  if (i >= NTOK * NTOK) return;
  int id = ridx[i];
  #pragma unroll
  for (int h = 0; h < NHEAD; ++h)
    bias[(size_t)h * NTOK * NTOK + i] = tbl[(size_t)id * NHEAD + h];
}

// ------------------------------------------------------------- layernorms
__global__ __launch_bounds__(128) void swin_ln_win(const float* __restrict__ x,
    const float* __restrict__ g, const float* __restrict__ b, u16* __restrict__ out){
  __shared__ float s[128];
  int row = blockIdx.x;
  int win = row / NTOK, n = row % NTOK;
  int dblk = win >> 6, hblk = (win >> 3) & 7, wblk = win & 7;
  int dz = n / 49, rem = n % 49, hy = rem / 7, wx = rem % 7;
  int d = (dblk*8 + dz + 4) & 15;
  int h = (hblk*7 + hy + 3) % 56;
  int w = (wblk*7 + wx + 3) % 56;
  size_t src = ((size_t)((d*56 + h)*56 + w)) * CDIM;
  int c = threadIdx.x;
  float v = x[src + c];
  s[c] = v; __syncthreads();
  for (int o = 64; o > 0; o >>= 1){ if (c < o) s[c] += s[c+o]; __syncthreads(); }
  float mean = s[0] * (1.0f/128.0f);
  __syncthreads();
  float dv = v - mean;
  s[c] = dv*dv; __syncthreads();
  for (int o = 64; o > 0; o >>= 1){ if (c < o) s[c] += s[c+o]; __syncthreads(); }
  float var = s[0] * (1.0f/128.0f);
  out[(size_t)row*CDIM + c] = f2bf(dv * rsqrtf(var + 1e-5f) * g[c] + b[c]);
}

__global__ __launch_bounds__(128) void swin_ln_lin(const float* __restrict__ x,
    const float* __restrict__ g, const float* __restrict__ b, u16* __restrict__ out){
  __shared__ float s[128];
  int row = blockIdx.x;
  int c = threadIdx.x;
  float v = x[(size_t)row*CDIM + c];
  s[c] = v; __syncthreads();
  for (int o = 64; o > 0; o >>= 1){ if (c < o) s[c] += s[c+o]; __syncthreads(); }
  float mean = s[0] * (1.0f/128.0f);
  __syncthreads();
  float dv = v - mean;
  s[c] = dv*dv; __syncthreads();
  for (int o = 64; o > 0; o >>= 1){ if (c < o) s[c] += s[c+o]; __syncthreads(); }
  float var = s[0] * (1.0f/128.0f);
  out[(size_t)row*CDIM + c] = f2bf(dv * rsqrtf(var + 1e-5f) * g[c] + b[c]);
}

// ------------------------------------------------------------- GEMM (bf16, WMMA)
// Block tile 128(M) x 64(N), 8 waves, wave = 16 rows x 4 accum tiles, K chunks of 32.
template <class Epi>
__global__ __launch_bounds__(256) void swin_gemm_bf16(const u16* __restrict__ A,
    const u16* __restrict__ Bw, int M, int Nn, int K, Epi epi){
  __shared__ u32 As[128*16];   // 128 rows x 32 K (bf16 pairs as dwords)
  __shared__ u32 Bs[64*16];    // 64 cols (transposed) x 32 K

  int tid = threadIdx.x, lane = tid & 31, wv = tid >> 5;
  int m0 = blockIdx.x * 128, n0 = blockIdx.y * 64;
  int r = lane & 15, g = lane >> 4;

  v8f acc[4];
  #pragma unroll
  for (int nt = 0; nt < 4; ++nt) acc[nt] = vzero();

  int nk = K >> 5;
  for (int kt = 0; kt < nk; ++kt){
    int k0 = kt * 32;
    // prefetch next K chunk into cache while this chunk is staged + computed
    if (kt + 1 < nk){
      int dwA = tid * 8;
      int rowA = dwA >> 4, kdA = dwA & 15;
      __builtin_prefetch(A + (size_t)(m0 + rowA)*K + k0 + 32 + kdA*2, 0, 3);
      int dwB = tid * 4;
      int nnB = dwB >> 4, kdB = dwB & 15;
      __builtin_prefetch(Bw + (size_t)(k0 + 32 + kdB*2)*Nn + n0 + nnB, 0, 3);
    }
    #pragma unroll
    for (int i = 0; i < 8; ++i){           // A: 2048 dwords
      int dw = tid*8 + i;
      int row = dw >> 4, kd = dw & 15;
      As[dw] = *(const u32*)(A + (size_t)(m0 + row)*K + k0 + kd*2);
    }
    #pragma unroll
    for (int i = 0; i < 4; ++i){           // B -> transposed: 1024 dwords
      int dw = tid*4 + i;
      int nn = dw >> 4, kd = dw & 15;
      int kk = k0 + kd*2;
      u32 e0 = Bw[(size_t)kk * Nn + n0 + nn];
      u32 e1 = Bw[(size_t)(kk+1) * Nn + n0 + nn];
      Bs[dw] = (e0 & 0xffffu) | (e1 << 16);
    }
    __syncthreads();
    Frag a;
    const u32* arow = &As[(wv*16 + r) * 16];
    #pragma unroll
    for (int j = 0; j < 8; ++j){
      int kd = (j < 4) ? (g*4 + j) : (8 + g*4 + (j-4));
      a.u[j] = arow[kd];
    }
    #pragma unroll
    for (int nt = 0; nt < 4; ++nt){
      Frag b;
      const u32* brow = &Bs[(nt*16 + r) * 16];
      #pragma unroll
      for (int j = 0; j < 8; ++j) b.u[j] = brow[(j < 4) ? (g*4 + j) : (8 + g*4 + (j-4))];
      acc[nt] = wmma_bf16(a, b, acc[nt]);
    }
    __syncthreads();
  }
  #pragma unroll
  for (int nt = 0; nt < 4; ++nt)
    #pragma unroll
    for (int i = 0; i < 8; ++i)
      epi(m0 + wv*16 + g*8 + i, n0 + nt*16 + r, acc[nt][i]);
}

// ------------------------------------------------------------- epilogues
struct EpiQKV {
  const float* bias; u16 *q, *k, *v;
  __device__ void operator()(int row, int col, float val) const {
    val += bias[col];
    int which = col >> 7, head = (col >> 5) & 3, dch = col & 31;
    int win = row / NTOK, n = row % NTOK;
    u16* dst = (which == 0) ? q : (which == 1) ? k : v;
    if (which == 0) val *= 0.1767766953f;          // 1/sqrt(32)
    dst[(((size_t)(win*NHEAD + head)) * NPAD + n) * HDIM + dch] = f2bf(val);
  }
};

struct EpiProj {   // window-reverse + roll-back + residual add -> x2 (f32, spatial)
  const float* bias; const float* xin; float* x2;
  __device__ void operator()(int row, int col, float val) const {
    val += bias[col];
    int win = row / NTOK, n = row % NTOK;
    int dblk = win >> 6, hblk = (win >> 3) & 7, wblk = win & 7;
    int dz = n / 49, rem = n % 49, hy = rem / 7, wx = rem % 7;
    int d = (dblk*8 + dz + 4) & 15;
    int h = (hblk*7 + hy + 3) % 56;
    int w = (wblk*7 + wx + 3) % 56;
    size_t idx = ((size_t)((d*56 + h)*56 + w)) * CDIM + col;
    x2[idx] = xin[idx] + val;
  }
};

struct EpiFC1 {    // exact GELU, bf16 hidden
  const float* bias; u16* h1;
  __device__ void operator()(int row, int col, float val) const {
    val += bias[col];
    float gel = 0.5f * val * (1.0f + erff(val * 0.70710678f));
    h1[(size_t)row * HIDD + col] = f2bf(gel);
  }
};

struct EpiFC2 {    // final residual, f32 output
  const float* bias; const float* x2; float* out;
  __device__ void operator()(int row, int col, float val) const {
    size_t idx = (size_t)row * CDIM + col;
    out[idx] = x2[idx] + val + bias[col];
  }
};

// ------------------------------------------------------------- attention
// grid.x = win*4+head (512), grid.y = 16-row query tile (25). 256 threads.
__global__ __launch_bounds__(256) void swin_attn(const u16* __restrict__ qb,
    const u16* __restrict__ kb, const u16* __restrict__ vb,
    const float* __restrict__ biasb, const float* __restrict__ mask,
    u16* __restrict__ attnout){
  __shared__ float sc[16][NPAD];                  // 26.0 KB
  __shared__ __align__(16) u16 pbf[16][NPAD];     // 13.0 KB
  __shared__ float red[16][16];
  __shared__ float rowstat[16];
  __shared__ float outacc[16][32];

  int wh = blockIdx.x;
  int win = wh >> 2, head = wh & 3;
  int n0 = blockIdx.y * 16;
  int tid = threadIdx.x, lane = tid & 31, wv = tid >> 5;
  int cc = lane & 15, g = lane >> 4;

  const u16* q0 = qb + (size_t)wh * NPAD * HDIM;
  const u16* k0 = kb + (size_t)wh * NPAD * HDIM;
  const u16* v0 = vb + (size_t)wh * NPAD * HDIM;
  const float* bp = biasb + (size_t)head * NTOK * NTOK;
  const float* mp = mask  + (size_t)win  * NTOK * NTOK;

  // ---- stage 1: scores = q k^T (+bias+mask), one WMMA per 16x16 tile (K=32)
  Frag a;
  {
    const u32* qrow = (const u32*)(q0 + (size_t)(n0 + cc) * HDIM);
    #pragma unroll
    for (int j = 0; j < 8; ++j){
      int kd = (j < 4) ? (g*4 + j) : (8 + g*4 + (j-4));
      a.u[j] = qrow[kd];
    }
  }
  for (int ct = wv; ct < 26; ct += 8){
    Frag b;
    const u32* krow = (const u32*)(k0 + (size_t)(ct*16 + cc) * HDIM);
    #pragma unroll
    for (int j = 0; j < 8; ++j) b.u[j] = krow[(j < 4) ? (g*4 + j) : (8 + g*4 + (j-4))];
    v8f av = vzero();
    av = wmma_bf16(a, b, av);
    int col = ct*16 + cc;
    int ms = (col < NTOK) ? col : (NTOK - 1);        // clamp, branch-free
    bool colok = (col < NTOK);
    #pragma unroll
    for (int i = 0; i < 8; ++i){
      int rr = i + g*8;
      int n = n0 + rr;
      int ns = (n < NTOK) ? n : (NTOK - 1);          // clamp, branch-free
      // unconditional loads (clamped addresses are always valid), then select
      float bm = bp[(size_t)ns*NTOK + ms] + mp[(size_t)ns*NTOK + ms];
      float addv = (colok && (n < NTOK)) ? bm : -1e30f;
      sc[rr][col] = av[i] + addv;
    }
  }
  __syncthreads();

  // ---- stage 2: row softmax over 416 cols (16 rows x 16 col-groups)
  int r = tid >> 4, cg = tid & 15;
  float mx = -1e30f;
  for (int c = cg; c < NPAD; c += 16) mx = fmaxf(mx, sc[r][c]);
  red[r][cg] = mx; __syncthreads();
  if (cg == 0){
    float m2 = red[r][0];
    #pragma unroll
    for (int i = 1; i < 16; ++i) m2 = fmaxf(m2, red[r][i]);
    rowstat[r] = m2;
  }
  __syncthreads();
  float rm = rowstat[r];
  float sum = 0.0f;
  for (int c = cg; c < NPAD; c += 16){
    float e = __expf(sc[r][c] - rm);
    sc[r][c] = e; sum += e;
  }
  red[r][cg] = sum; __syncthreads();
  if (cg == 0){
    float s2 = 0.0f;
    #pragma unroll
    for (int i = 0; i < 16; ++i) s2 += red[r][i];
    rowstat[r] = 1.0f / s2;
  }
  __syncthreads();
  float inv = rowstat[r];
  for (int c = cg; c < NPAD; c += 16) pbf[r][c] = f2bf(sc[r][c] * inv);
  for (int i = tid; i < 512; i += 256) ((float*)outacc)[i] = 0.0f;
  __syncthreads();

  // ---- stage 3: out = P @ v, K(=416) split across waves, ds-atomic reduce
  v8f acc0 = vzero(), acc1 = vzero();
  for (int kc = wv; kc < 13; kc += 8){
    Frag pa;
    const u32* prow = (const u32*)&pbf[cc][0];
    #pragma unroll
    for (int j = 0; j < 8; ++j){
      int kd = (j < 4) ? (g*4 + j) : (8 + g*4 + (j-4));
      pa.u[j] = prow[kc*16 + kd];
    }
    #pragma unroll
    for (int nt = 0; nt < 2; ++nt){
      Frag b;
      #pragma unroll
      for (int j = 0; j < 8; ++j){
        int kk = kc*32 + ((j < 4) ? (g*8 + 2*j) : (16 + g*8 + 2*(j-4)));
        u32 e0 = v0[(size_t)kk * HDIM + nt*16 + cc];
        u32 e1 = v0[(size_t)(kk+1) * HDIM + nt*16 + cc];
        b.u[j] = (e0 & 0xffffu) | (e1 << 16);
      }
      if (nt == 0) acc0 = wmma_bf16(pa, b, acc0);
      else         acc1 = wmma_bf16(pa, b, acc1);
    }
  }
  #pragma unroll
  for (int i = 0; i < 8; ++i){
    int rr = i + g*8;
    atomicAdd(&outacc[rr][cc],      acc0[i]);
    atomicAdd(&outacc[rr][16 + cc], acc1[i]);
  }
  __syncthreads();
  for (int i = tid; i < 512; i += 256){
    int rr = i >> 5, c = i & 31;
    int n = n0 + rr;
    if (n < NTOK)
      attnout[((size_t)(win*NTOK + n)) * CDIM + head*HDIM + c] = f2bf(outacc[rr][c]);
  }
}

// ------------------------------------------------------------- launcher
extern "C" void kernel_launch(void* const* d_in, const int* in_sizes, int n_in,
                              void* d_out, int out_size, void* d_ws, size_t ws_size,
                              hipStream_t stream){
  const float* x       = (const float*)d_in[0];
  const float* mask    = (const float*)d_in[1];
  const float* n1g     = (const float*)d_in[2];
  const float* n1b     = (const float*)d_in[3];
  const float* qkv_w   = (const float*)d_in[4];
  const float* qkv_b   = (const float*)d_in[5];
  const float* proj_w  = (const float*)d_in[6];
  const float* proj_b  = (const float*)d_in[7];
  const float* rel_tbl = (const float*)d_in[8];
  const float* n2g     = (const float*)d_in[9];
  const float* n2b     = (const float*)d_in[10];
  const float* fc1_w   = (const float*)d_in[11];
  const float* fc1_b   = (const float*)d_in[12];
  const float* fc2_w   = (const float*)d_in[13];
  const float* fc2_b   = (const float*)d_in[14];
  const int*   rel_idx = (const int*)d_in[15];
  float* out = (float*)d_out;

  char* ws = (char*)d_ws;
  u16*   qkvw_bf = (u16*)(ws + 0);          // 98304 B
  u16*   projw_bf= (u16*)(ws + 98304);      // 32768 B
  u16*   fc1w_bf = (u16*)(ws + 131072);     // 131072 B
  u16*   fc2w_bf = (u16*)(ws + 262144);     // 131072 B
  float* biasb   = (float*)(ws + 393216);   // 2458624 B
  float* x2      = (float*)(ws + 2851840);  // 25690112 B (persistent)
  u16*   xln     = (u16*)(ws + 28541952);   // 12845056 B (reused as xln2)
  u16*   qb      = (u16*)(ws + 41387008);   // 13631488 B
  u16*   kb      = (u16*)(ws + 55018496);   // 13631488 B
  u16*   vb      = (u16*)(ws + 68649984);   // 13631488 B
  u16*   attnout = (u16*)(ws + 82281472);   // 12845056 B
  u16*   h1      = (u16*)(ws + 41387008);   // reuse q/k/v region after attention

  // weights -> bf16, bias table, zero padded q/k/v
  swin_cvt_bf16<<<192, 256, 0, stream>>>(qkv_w,  qkvw_bf,  (long)128*384);
  swin_cvt_bf16<<<64,  256, 0, stream>>>(proj_w, projw_bf, (long)128*128);
  swin_cvt_bf16<<<256, 256, 0, stream>>>(fc1_w,  fc1w_bf,  (long)128*512);
  swin_cvt_bf16<<<256, 256, 0, stream>>>(fc2_w,  fc2w_bf,  (long)512*128);
  swin_bias_kernel<<<(NTOK*NTOK + 255)/256, 256, 0, stream>>>(rel_tbl, rel_idx, biasb);
  swin_zero_u32<<<2048, 256, 0, stream>>>((u32*)qb, (long)(3L*13631488/4));

  // LN1 (+roll+window partition) -> bf16
  swin_ln_win<<<MTOT, 128, 0, stream>>>(x, n1g, n1b, xln);

  // QKV GEMM: (50176x128)@(128x384) -> q/k/v (padded, q pre-scaled)
  swin_gemm_bf16<EpiQKV><<<dim3(392, 6), 256, 0, stream>>>(
      xln, qkvw_bf, MTOT, 384, 128, EpiQKV{qkv_b, qb, kb, vb});

  // attention: 128 windows x 4 heads x 25 query tiles
  swin_attn<<<dim3(NWIN*NHEAD, 25), 256, 0, stream>>>(qb, kb, vb, biasb, mask, attnout);

  // proj GEMM + window-reverse + residual -> x2 (f32 spatial)
  swin_gemm_bf16<EpiProj><<<dim3(392, 2), 256, 0, stream>>>(
      attnout, projw_bf, MTOT, 128, 128, EpiProj{proj_b, x, x2});

  // LN2 -> bf16 (reuse xln buffer)
  swin_ln_lin<<<MTOT, 128, 0, stream>>>(x2, n2g, n2b, xln);

  // MLP: fc1+gelu, fc2+residual
  swin_gemm_bf16<EpiFC1><<<dim3(392, 8), 256, 0, stream>>>(
      xln, fc1w_bf, MTOT, HIDD, 128, EpiFC1{fc1_b, h1});
  swin_gemm_bf16<EpiFC2><<<dim3(392, 2), 256, 0, stream>>>(
      h1, fc2w_bf, MTOT, 128, HIDD, EpiFC2{fc2_b, x2, out});
}
